// ConvFourierKANLayer_30159260352767
// MI455X (gfx1250) — matrix operand order repaired and checked by reading, not compiled, not run on hardware
//
#include <hip/hip_runtime.h>
#include <hip/hip_bf16.h>

// ---------------------------------------------------------------------------
// ConvFourierKANLayer for MI455X (gfx1250, wave32, WMMA, async-to-LDS)
//
//   B=16, C=64, H=W=64, G=10, O=128, 3x3 conv with pad=1
//   GEMM view: M=O=128, N=B*H*W=65536, K=2*C*G*9=11520, f16 x f16 -> f32 acc
// ---------------------------------------------------------------------------

typedef _Float16 h8  __attribute__((ext_vector_type(8)));
typedef _Float16 v16h __attribute__((ext_vector_type(16)));
typedef float    v8f  __attribute__((ext_vector_type(8)));

#define BB   16
#define CC   64
#define HH   64
#define WW   64
#define GG   10
#define OO   128
#define KCH  (2 * CC * GG)   // 1280 f16 channels per pixel (cos/sin interleaved)
#define KC_STEP 64           // K chunk staged per stage
#define NSTAGE (9 * (KCH / KC_STEP))   // 9 taps x 20 chunks = 180 stages
#define LPITCH 72            // LDS row pitch in f16 (144B -> conflict-free)

// LDS arena layout (f16 units); single __shared__ array => base LDS offset 0
#define LDS_A0  0
#define LDS_A1  (OO * LPITCH)                    //  9216
#define LDS_B0  (2 * OO * LPITCH)                // 18432
#define LDS_B1  (2 * OO * LPITCH + 264 * LPITCH) // 37440
#define LDS_TOT (2 * OO * LPITCH + 2 * 264 * LPITCH) // 56448 f16 = 112896 B

// ---------------------------------------------------------------------------
// Kernel 0: repack fourier coeffs (2,O,C,3,3,G) f32 -> f16 [tap][o][kc]
//           kc = c*20 + g*2 + t   (t: 0=cos branch, 1=sin branch)
// ---------------------------------------------------------------------------
__global__ __launch_bounds__(256) void kan_repack_w(const float* __restrict__ fc,
                                                    _Float16* __restrict__ wbuf) {
    int idx = blockIdx.x * 256 + threadIdx.x;          // (tap*O + o)*KCH + kc
    const int total = 9 * OO * KCH;
    if (idx >= total) return;
    int kc  = idx % KCH;
    int rem = idx / KCH;
    int o   = rem % OO;
    int tap = rem / OO;
    int kh = tap / 3, kw = tap % 3;
    int c = kc / 20;
    int r2 = kc % 20;
    int g = r2 >> 1;
    int t = r2 & 1;
    size_t src = ((((size_t)t * OO + o) * CC + c) * 3 + kh) * 3 + kw;
    src = src * GG + g;
    wbuf[idx] = (_Float16)fc[src];
}

// ---------------------------------------------------------------------------
// Kernel 1: featurize  x:(B,C,H,W) f32 -> feat[pix][kc] f16
// ---------------------------------------------------------------------------
__global__ __launch_bounds__(256) void kan_featurize(const float* __restrict__ x,
                                                     _Float16* __restrict__ feat) {
    int idx = blockIdx.x * 256 + threadIdx.x;          // pix*64 + c
    const int total = BB * HH * WW * CC;
    if (idx >= total) return;
    int c   = idx & (CC - 1);
    int pix = idx >> 6;
    int w = pix & (WW - 1);
    int h = (pix >> 6) & (HH - 1);
    int b = pix >> 12;
    float v = x[(((size_t)b * CC + c) * HH + h) * WW + w];
    unsigned* dst = (unsigned*)(feat + (size_t)pix * KCH + c * 20);
    #pragma unroll
    for (int g = 0; g < GG; ++g) {
        float s, co;
        __sincosf(v * (float)(g + 1), &s, &co);
        union { _Float16 hh2[2]; unsigned u; } p;
        p.hh2[0] = (_Float16)co;
        p.hh2[1] = (_Float16)s;
        dst[g] = p.u;
    }
}

// ---------------------------------------------------------------------------
// Async 16B copy global -> LDS (gfx1250), SADDR form, tracked by ASYNCcnt
// ---------------------------------------------------------------------------
__device__ __forceinline__ void async_copy16(unsigned lds_byte, unsigned gbyte_off,
                                             const void* base) {
    asm volatile("global_load_async_to_lds_b128 %0, %1, %2"
                 :: "v"(lds_byte), "v"(gbyte_off), "s"(base)
                 : "memory");
}

__device__ __forceinline__ void wait_async0() {
    asm volatile("s_wait_asynccnt 0" ::: "memory");
}

// ---------------------------------------------------------------------------
// Fragment loaders following CDNA5 ISA 7.12.2 VGPR layouts (wave32)
// ---------------------------------------------------------------------------
__device__ __forceinline__ v16h load_afrag(const _Float16* row, int khalf) {
    union { v16h v; h8 h[2]; } u;
    u.h[0] = *(const h8*)(row + khalf * 8);
    u.h[1] = *(const h8*)(row + 16 + khalf * 8);
    return u.v;
}

__device__ __forceinline__ v16h load_bfrag(const _Float16* row, int khalf) {
    union { v16h v; h8 h[2]; } u;
    u.h[0] = *(const h8*)(row + khalf * 8);
    u.h[1] = *(const h8*)(row + 16 + khalf * 8);
    return u.v;
}

// ---------------------------------------------------------------------------
// Kernel 2: implicit-GEMM conv, double-buffered async-to-LDS pipeline.
// Block = 256 thr = 8 waves. Tile: M=128 (all O) x N=128 pixels (2 rows x 64).
// ---------------------------------------------------------------------------
__global__ __launch_bounds__(256) void kan_gemm(const _Float16* __restrict__ wbuf,
                                                const _Float16* __restrict__ feat,
                                                const float* __restrict__ bias,
                                                float* __restrict__ out) {
    __shared__ _Float16 smem[LDS_TOT];

    const int tid  = threadIdx.x;
    const int blk  = blockIdx.x;                  // 512 blocks
    const int b    = blk >> 5;                    // image
    const int h0   = (blk & 31) * 2;              // first of 2 output rows
    const int wave = tid >> 5;
    const int lane = tid & 31;
    const int l16  = lane & 15;
    const int khalf = lane >> 4;
    const int wm = wave >> 2;                     // 0..1 -> 64 O rows
    const int wn = wave & 3;                      // 0..3 -> 32 pixels

    // ---- pre-zero both B halo buffers once (padding stays zero forever) ----
    for (int i = tid; i < (2 * 264 * LPITCH) / 8; i += 256) {
        *(uint4*)(smem + LDS_B0 + i * 8) = make_uint4(0u, 0u, 0u, 0u);
    }

    v8f acc[4][2];
    #pragma unroll
    for (int mi = 0; mi < 4; ++mi)
        #pragma unroll
        for (int ni = 0; ni < 2; ++ni)
            acc[mi][ni] = 0.0f;

    // stage issue: A slab always; B halo when tap==0 of that stage
    auto issue_stage = [&](int tap, int kcc, int parity) {
        const int kc = kcc * KC_STEP;
        const unsigned abase = (parity ? LDS_A1 : LDS_A0) * 2u;  // bytes
        #pragma unroll
        for (int i = 0; i < 4; ++i) {            // 1024 chunks / 256 threads
            int idx = tid + i * 256;
            int o = idx >> 3, sub = idx & 7;
            unsigned lds = abase + (unsigned)(o * LPITCH + sub * 8) * 2u;
            unsigned gof = (unsigned)(((tap * OO + o) * KCH + kc + sub * 8) * 2);
            async_copy16(lds, gof, wbuf);
        }
        if (tap == 0) {
            const unsigned bbase = ((kcc & 1) ? LDS_B1 : LDS_B0) * 2u;
            for (int i = tid; i < 264 * 8; i += 256) {
                int hp = i >> 3, sub = i & 7;
                int hh = hp / 66, ww = hp % 66;
                int h = h0 + hh - 1, w = ww - 1;
                if ((unsigned)h < (unsigned)HH && (unsigned)w < (unsigned)WW) {
                    unsigned lds = bbase + (unsigned)(hp * LPITCH + sub * 8) * 2u;
                    unsigned gof =
                        (unsigned)((((b * HH + h) * WW + w) * KCH + kc + sub * 8) * 2);
                    async_copy16(lds, gof, feat);
                }
            }
        }
    };

    // B zero-fill must land before first async halo writes complete ordering:
    __syncthreads();

    // ---- prologue: stage 0 ----
    issue_stage(/*tap=*/0, /*kcc=*/0, /*parity=*/0);
    wait_async0();
    __syncthreads();

    int tap = 0, kcc = 0;
    for (int s = 0; s < NSTAGE; ++s) {
        int ntap = tap + 1, nkcc = kcc;
        if (ntap == 9) { ntap = 0; nkcc = kcc + 1; }
        if (s + 1 < NSTAGE)
            issue_stage(ntap, nkcc, (s + 1) & 1);

        // ---- compute stage s ----
        const int kh = tap / 3, kw = tap - (tap / 3) * 3;
        const _Float16* Ab = smem + ((s & 1) ? LDS_A1 : LDS_A0);
        const _Float16* Bb = smem + ((kcc & 1) ? LDS_B1 : LDS_B0);
        #pragma unroll
        for (int kk = 0; kk < 2; ++kk) {          // two K=32 steps per 64-chunk
            v16h afrag[4], bfrag[2];
            #pragma unroll
            for (int mi = 0; mi < 4; ++mi)
                afrag[mi] = load_afrag(
                    Ab + (wm * 64 + mi * 16 + l16) * LPITCH + kk * 32, khalf);
            #pragma unroll
            for (int ni = 0; ni < 2; ++ni) {
                int n = (wn * 2 + ni) * 16 + l16;          // 0..127
                int r = n >> 6, w = n & 63;
                int hp = (r + kh) * 66 + (w + kw);
                bfrag[ni] = load_bfrag(Bb + hp * LPITCH + kk * 32, khalf);
            }
            #pragma unroll
            for (int mi = 0; mi < 4; ++mi)
                #pragma unroll
                for (int ni = 0; ni < 2; ++ni)
                    acc[mi][ni] = __builtin_amdgcn_wmma_f32_16x16x32_f16(
                        false, afrag[mi], false, bfrag[ni],
                        (short)0, acc[mi][ni], false, false);
        }

        wait_async0();
        __syncthreads();
        tap = ntap; kcc = nkcc;
    }

    // ---- epilogue: C/D layout M = r + 8*khalf, N = l16 ----
    #pragma unroll
    for (int mi = 0; mi < 4; ++mi) {
        #pragma unroll
        for (int ni = 0; ni < 2; ++ni) {
            int n = (wn * 2 + ni) * 16 + l16;
            int rr = n >> 6, w = n & 63;
            int h = h0 + rr;
            #pragma unroll
            for (int r = 0; r < 8; ++r) {
                int o = wm * 64 + mi * 16 + khalf * 8 + r;
                out[(((size_t)b * OO + o) * HH + h) * WW + w] = acc[mi][ni][r] + bias[o];
            }
        }
    }
}

// ---------------------------------------------------------------------------
extern "C" void kernel_launch(void* const* d_in, const int* in_sizes, int n_in,
                              void* d_out, int out_size, void* d_ws, size_t ws_size,
                              hipStream_t stream) {
    const float* x    = (const float*)d_in[0];   // (16,64,64,64)
    const float* fc   = (const float*)d_in[1];   // (2,128,64,3,3,10)
    const float* bias = (const float*)d_in[2];   // (1,128,1,1)
    float* out = (float*)d_out;                  // (16,128,64,64)

    _Float16* wbuf = (_Float16*)d_ws;                       // 9*128*1280 f16 = 2.95 MB
    _Float16* feat = wbuf + (size_t)9 * OO * KCH;           // 65536*1280 f16 = 167.8 MB
    (void)ws_size; (void)in_sizes; (void)n_in; (void)out_size;

    {   // weight repack: 1,474,560 elems
        int total = 9 * OO * KCH;
        kan_repack_w<<<(total + 255) / 256, 256, 0, stream>>>(fc, wbuf);
    }
    {   // featurize: 4,194,304 threads
        int total = BB * HH * WW * CC;
        kan_featurize<<<(total + 255) / 256, 256, 0, stream>>>(x, feat);
    }
    {   // implicit GEMM conv: 512 blocks x 256 threads
        kan_gemm<<<BB * (HH / 2), 256, 0, stream>>>(wbuf, feat, bias, out);
    }
}
